// REPL_27608049778742
// MI455X (gfx1250) — matrix operand order repaired and checked by reading, not compiled, not run on hardware
//
#include <hip/hip_runtime.h>

// ---------------------------------------------------------------------------
// Differential attention block for MI455X (gfx1250, wave32, WMMA bf16,
// async Global->LDS double-buffered staging).
//   B=2, T=2048, D=1024, H=16 heads, per-half head dim h=32
// ---------------------------------------------------------------------------

#define LAMBDA_INIT 0.35550906759096926f   // 0.8 - 0.6*exp(-0.3)

typedef __attribute__((ext_vector_type(16))) __bf16 bf16x16;
typedef __attribute__((ext_vector_type(8)))  __bf16 bf16x8;
typedef __attribute__((ext_vector_type(8)))  float  f32x8;

static __device__ __forceinline__ __bf16 f2bf(float f) {
  unsigned u = __float_as_uint(f);
  u += 0x7FFFu + ((u >> 16) & 1u);            // round-to-nearest-even
  unsigned short h = (unsigned short)(u >> 16);
  __bf16 b;
  __builtin_memcpy(&b, &h, sizeof(h));
  return b;
}

// Async copy of 16 bytes Global -> LDS (GVS mode, tracked by ASYNCcnt).
// lds_off: wave-relative LDS byte address (generic LDS ptr truncated to 32b).
// voff:    per-lane unsigned byte offset added to the SGPR base.
static __device__ __forceinline__ void async_b128(unsigned lds_off,
                                                  const void* base,
                                                  unsigned voff) {
  asm volatile("global_load_async_to_lds_b128 %0, %1, %2"
               :: "v"(lds_off), "v"(voff), "s"(base)
               : "memory");
}
#define WAIT_ASYNC(imm) asm volatile("s_wait_asynccnt " #imm ::: "memory")

static __device__ __forceinline__ unsigned lds_addr(const void* p) {
  return (unsigned)(unsigned long long)p;   // LDS aperture: addr[31:0]
}

// A-fragment (16x32 bf16): lane holds row = lane&15.
// elems 0..7 = K(half*8+0..7), elems 8..15 = K(16+half*8+0..7)
static __device__ __forceinline__ bf16x16 make_afrag(const __bf16* rowp, int half) {
  bf16x8 lo = *(const bf16x8*)(rowp + half * 8);
  bf16x8 hi = *(const bf16x8*)(rowp + 16 + half * 8);
  bf16x16 r;
#pragma unroll
  for (int i = 0; i < 8; ++i) { r[i] = lo[i]; r[8 + i] = hi[i]; }
  return r;
}

// B-fragment (32x16 bf16), source N-major: lane holds col = lane&15,
// elems 0..7 = K(half*8+0..7), elems 8..15 = K(16+half*8+0..7)
static __device__ __forceinline__ bf16x16 make_bfrag(const __bf16* colp, int half) {
  bf16x8 lo = *(const bf16x8*)(colp + half * 8);
  bf16x8 hi = *(const bf16x8*)(colp + 16 + half * 8);
  bf16x16 r;
#pragma unroll
  for (int i = 0; i < 8; ++i) { r[i] = lo[i]; r[8 + i] = hi[i]; }
  return r;
}

static __device__ __forceinline__ f32x8 wmma_bf16(bf16x16 a, bf16x16 b, f32x8 c) {
  return __builtin_amdgcn_wmma_f32_16x16x32_bf16(false, a, false, b, (short)0, c,
                                                 false, false);
}

static __device__ __forceinline__ f32x8 zero8() {
  f32x8 z;
#pragma unroll
  for (int e = 0; e < 8; ++e) z[e] = 0.f;
  return z;
}

// ---------------------------------------------------------------------------
// Converters
// ---------------------------------------------------------------------------
__global__ void cvt_f32_bf16(const float* __restrict__ in, __bf16* __restrict__ out,
                             int n) {
  int i = blockIdx.x * blockDim.x + threadIdx.x;
  if (i < n) out[i] = f2bf(in[i]);
}

// in [rows][cols] fp32 -> out [cols][rows] bf16  (N-major weights)
__global__ void cvt_transpose_bf16(const float* __restrict__ in,
                                   __bf16* __restrict__ out, int rows, int cols) {
  int i = blockIdx.x * blockDim.x + threadIdx.x;
  if (i < rows * cols) {
    int r = i / cols, c = i % cols;
    out[(size_t)c * rows + r] = f2bf(in[i]);
  }
}

// ---------------------------------------------------------------------------
// QKV GEMM: A[4096][1024] bf16 (x) * BT[3072][1024] bf16 (Wqkv^T, N-major)
// Async double-buffered staging. Uniform-branch scatter epilogue into:
//   qb [B,2H,T,32], kb [B,2H,T,32], vtb [B,H,64,T] (V pre-transposed)
// ---------------------------------------------------------------------------
__global__ __launch_bounds__(256)
void gemm_qkv_kernel(const __bf16* __restrict__ A, const __bf16* __restrict__ BT,
                     __bf16* __restrict__ qb, __bf16* __restrict__ kb,
                     __bf16* __restrict__ vtb) {
  __shared__ __bf16 As[2][128 * 40];
  __shared__ __bf16 Bs[2][128 * 40];

  const int tid  = threadIdx.x;
  const int lane = tid & 31;
  const int w    = tid >> 5;
  const int wr   = w >> 1;        // 0..3 : 32-row group
  const int wc   = w & 1;         // 0..1 : 64-col group
  const int half = lane >> 4;
  const int l16  = lane & 15;
  const int m0   = blockIdx.y * 128;
  const int n0   = blockIdx.x * 128;
  const int srow = tid >> 1;      // 0..127
  const int spar = tid & 1;

  f32x8 acc[2][4];
#pragma unroll
  for (int i = 0; i < 2; ++i)
#pragma unroll
    for (int j = 0; j < 4; ++j) acc[i][j] = zero8();

  // 4 async b128 per thread per tile
  auto stage = [&](int k0, int buf) {
    const unsigned ga = (unsigned)(((m0 + srow) * 1024 + k0 + spar * 16) * 2);
    const unsigned la = lds_addr(&As[buf][srow * 40 + spar * 16]);
    async_b128(la, A, ga);
    async_b128(la + 16, A, ga + 16);
    const unsigned gb = (unsigned)(((n0 + srow) * 1024 + k0 + spar * 16) * 2);
    const unsigned lb = lds_addr(&Bs[buf][srow * 40 + spar * 16]);
    async_b128(lb, BT, gb);
    async_b128(lb + 16, BT, gb + 16);
  };

  stage(0, 0);
  for (int t = 0; t < 32; ++t) {
    const int buf = t & 1;
    if (t < 31) {
      stage((t + 1) * 32, buf ^ 1);
      WAIT_ASYNC(0x4);            // current tile's 4 ops are done (in-order)
    } else {
      WAIT_ASYNC(0x0);
    }
    __syncthreads();

    bf16x16 af[2], bfr[4];
#pragma unroll
    for (int i = 0; i < 2; ++i)
      af[i] = make_afrag(&As[buf][(wr * 32 + i * 16 + l16) * 40], half);
#pragma unroll
    for (int j = 0; j < 4; ++j)
      bfr[j] = make_bfrag(&Bs[buf][(wc * 64 + j * 16 + l16) * 40], half);
#pragma unroll
    for (int i = 0; i < 2; ++i)
#pragma unroll
      for (int j = 0; j < 4; ++j) acc[i][j] = wmma_bf16(af[i], bfr[j], acc[i][j]);
    __syncthreads();
  }

  // Uniform-region scatter epilogue (BN=128 divides the 1024/2048 boundaries).
  if (n0 < 1024) {                      // ---- Q region
#pragma unroll
    for (int i = 0; i < 2; ++i)
#pragma unroll
      for (int j = 0; j < 4; ++j) {
        const int n = n0 + wc * 64 + j * 16 + l16;
        const int hh = n >> 5, d = n & 31;
#pragma unroll
        for (int r = 0; r < 8; ++r) {
          const int m = m0 + wr * 32 + i * 16 + half * 8 + r;
          const int b = m >> 11, t = m & 2047;
          qb[((size_t)((b * 32 + hh) * 2048 + t)) * 32 + d] = f2bf(acc[i][j][r]);
        }
      }
  } else if (n0 < 2048) {               // ---- K region
#pragma unroll
    for (int i = 0; i < 2; ++i)
#pragma unroll
      for (int j = 0; j < 4; ++j) {
        const int nn = n0 - 1024 + wc * 64 + j * 16 + l16;
        const int hh = nn >> 5, d = nn & 31;
#pragma unroll
        for (int r = 0; r < 8; ++r) {
          const int m = m0 + wr * 32 + i * 16 + half * 8 + r;
          const int b = m >> 11, t = m & 2047;
          kb[((size_t)((b * 32 + hh) * 2048 + t)) * 32 + d] = f2bf(acc[i][j][r]);
        }
      }
  } else {                              // ---- V region (store transposed)
#pragma unroll
    for (int i = 0; i < 2; ++i)
#pragma unroll
      for (int j = 0; j < 4; ++j) {
        const int nn = n0 - 2048 + wc * 64 + j * 16 + l16;
        const int hh = nn >> 6, d = nn & 63;
#pragma unroll
        for (int r = 0; r < 8; ++r) {
          const int m = m0 + wr * 32 + i * 16 + half * 8 + r;
          const int b = m >> 11, t = m & 2047;
          vtb[((size_t)((b * 16 + hh) * 64 + d)) * 2048 + t] = f2bf(acc[i][j][r]);
        }
      }
  }
}

// ---------------------------------------------------------------------------
// Flash-style differential attention (one (b, head) x 64 Q-rows per block).
// 4 waves x 16 Q rows; 64-key tiles; both halves fused; async K/V staging.
// ---------------------------------------------------------------------------
static __device__ __forceinline__ void attn_step(
    const bf16x16& qf, const __bf16* Ks, const __bf16* Vsh, __bf16* Pw,
    float* ms, float* ls, f32x8* o, int half, int l16) {
  f32x8 s[4];
#pragma unroll
  for (int j = 0; j < 4; ++j) {
    bf16x16 bfr = make_bfrag(Ks + (j * 16 + l16) * 40, half);
    s[j] = wmma_bf16(qf, bfr, zero8());
  }
  const float sc = 0.17677669529663687f;  // 1/sqrt(32)
#pragma unroll
  for (int r = 0; r < 8; ++r) {
    float v = fmaxf(fmaxf(s[0][r], s[1][r]), fmaxf(s[2][r], s[3][r])) * sc;
#pragma unroll
    for (int sh = 1; sh < 16; sh <<= 1) v = fmaxf(v, __shfl_xor(v, sh, 32));
    const float mnew = fmaxf(ms[r], v);
    const float corr = __expf(ms[r] - mnew);
    float rowsum = 0.f;
#pragma unroll
    for (int j = 0; j < 4; ++j) {
      const float p = __expf(s[j][r] * sc - mnew);
      rowsum += p;
      Pw[(half * 8 + r) * 72 + j * 16 + l16] = f2bf(p);
    }
#pragma unroll
    for (int sh = 1; sh < 16; sh <<= 1) rowsum += __shfl_xor(rowsum, sh, 32);
    ms[r] = mnew;
    ls[r] = ls[r] * corr + rowsum;
#pragma unroll
    for (int j = 0; j < 4; ++j) o[j][r] *= corr;
  }
  // P (16x64) @ V (64x64): two K=32 steps, 4 N-tiles (same-wave LDS is in-order).
#pragma unroll
  for (int kk = 0; kk < 2; ++kk) {
    bf16x16 af = make_afrag(Pw + l16 * 72 + kk * 32, half);
#pragma unroll
    for (int j = 0; j < 4; ++j) {
      bf16x16 bfr = make_bfrag(Vsh + (j * 16 + l16) * 72 + kk * 32, half);
      o[j] = wmma_bf16(af, bfr, o[j]);
    }
  }
}

__global__ __launch_bounds__(128)
void diff_attn_kernel(const __bf16* __restrict__ qb, const __bf16* __restrict__ kb,
                      const __bf16* __restrict__ vtb,
                      const float* __restrict__ lq1, const float* __restrict__ lk1,
                      const float* __restrict__ lq2, const float* __restrict__ lk2,
                      const float* __restrict__ rms_scale,
                      __bf16* __restrict__ yb) {
  __shared__ __bf16 K1s[2][64 * 40];
  __shared__ __bf16 K2s[2][64 * 40];
  __shared__ __bf16 Vs[2][64 * 72];   // V transposed tile: [dim][key]
  __shared__ __bf16 Ps[4 * 16 * 72];  // per-wave P bounce buffer

  const int tid  = threadIdx.x;
  const int lane = tid & 31;
  const int w    = tid >> 5;  // 0..3
  const int half = lane >> 4;
  const int l16  = lane & 15;
  const int bh   = blockIdx.y;
  const int b    = bh >> 4, head = bh & 15;
  const int q0   = blockIdx.x * 64;
  const int srow = tid >> 1;  // 0..63
  const int spar = tid & 1;

  // lambda = exp(lq1.lk1) - exp(lq2.lk2) + LAMBDA_INIT  (h = 32 = wave width)
  float p1 = lq1[lane] * lk1[lane];
  float p2 = lq2[lane] * lk2[lane];
#pragma unroll
  for (int sh = 16; sh > 0; sh >>= 1) {
    p1 += __shfl_xor(p1, sh, 32);
    p2 += __shfl_xor(p2, sh, 32);
  }
  const float lam = __expf(p1) - __expf(p2) + LAMBDA_INIT;

  const __bf16* q1p = qb + ((size_t)(b * 32 + head) * 2048) * 32;
  const __bf16* q2p = qb + ((size_t)(b * 32 + 16 + head) * 2048) * 32;
  const __bf16* k1p = kb + ((size_t)(b * 32 + head) * 2048) * 32;
  const __bf16* k2p = kb + ((size_t)(b * 32 + 16 + head) * 2048) * 32;
  const __bf16* vp  = vtb + ((size_t)(b * 16 + head) * 64) * 2048;  // [64][2048]

  const int qrow = q0 + w * 16 + l16;
  const bf16x16 q1f = make_afrag(q1p + (size_t)qrow * 32, half);
  const bf16x16 q2f = make_afrag(q2p + (size_t)qrow * 32, half);

  float m1[8], l1[8], m2[8], l2[8];
  f32x8 o1[4], o2[4];
#pragma unroll
  for (int r = 0; r < 8; ++r) { m1[r] = -1e30f; l1[r] = 0.f; m2[r] = -1e30f; l2[r] = 0.f; }
#pragma unroll
  for (int j = 0; j < 4; ++j) { o1[j] = zero8(); o2[j] = zero8(); }

  __bf16* Pw = Ps + w * 16 * 72;

  // 8 async b128 per thread per 64-key tile
  auto stage = [&](int kt, int buf) {
    const unsigned gk = (unsigned)((((size_t)(kt + srow)) * 32 + spar * 16) * 2);
    const unsigned la1 = lds_addr(&K1s[buf][srow * 40 + spar * 16]);
    async_b128(la1, k1p, gk);
    async_b128(la1 + 16, k1p, gk + 16);
    const unsigned la2 = lds_addr(&K2s[buf][srow * 40 + spar * 16]);
    async_b128(la2, k2p, gk);
    async_b128(la2 + 16, k2p, gk + 16);
    const unsigned gv = (unsigned)(((size_t)srow * 2048 + kt + spar * 32) * 2);
    const unsigned lv = lds_addr(&Vs[buf][srow * 72 + spar * 32]);
#pragma unroll
    for (int c = 0; c < 4; ++c) async_b128(lv + c * 16, vp, gv + c * 16);
  };

  stage(0, 0);
  for (int t = 0; t < 32; ++t) {
    const int buf = t & 1;
    if (t < 31) {
      stage((t + 1) * 64, buf ^ 1);
      WAIT_ASYNC(0x8);            // current tile's 8 ops complete (in-order)
    } else {
      WAIT_ASYNC(0x0);
    }
    __syncthreads();

    attn_step(q1f, K1s[buf], Vs[buf], Pw, m1, l1, o1, half, l16);
    attn_step(q2f, K2s[buf], Vs[buf], Pw, m2, l2, o2, half, l16);

    __syncthreads();
  }

  // Finalize: diff, RMSNorm over 64 dims, scale, store bf16 y[B,T,D].
  const float oscale = 1.0f - LAMBDA_INIT;
  float outv[4][8];
  float rn[8];
#pragma unroll
  for (int r = 0; r < 8; ++r) {
    const float i1 = 1.f / l1[r];
    const float i2 = 1.f / l2[r];
    float ss = 0.f;
#pragma unroll
    for (int j = 0; j < 4; ++j) {
      const float v = o1[j][r] * i1 - lam * (o2[j][r] * i2);
      outv[j][r] = v;
      ss += v * v;
    }
#pragma unroll
    for (int sh = 1; sh < 16; sh <<= 1) ss += __shfl_xor(ss, sh, 32);
    rn[r] = rsqrtf(ss * (1.0f / 64.0f) + 1e-6f);
  }
#pragma unroll
  for (int j = 0; j < 4; ++j) {
    const float rsc = rms_scale[j * 16 + l16] * oscale;
#pragma unroll
    for (int r = 0; r < 8; ++r) {
      const int t = q0 + w * 16 + half * 8 + r;
      const float v = outv[j][r] * rn[r] * rsc;
      yb[(size_t)(b * 2048 + t) * 1024 + head * 64 + j * 16 + l16] = f2bf(v);
    }
  }
}

// ---------------------------------------------------------------------------
// Output GEMM: yb[4096][1024] bf16 * WprojT[1024][1024] bf16 -> out fp32
// ---------------------------------------------------------------------------
__global__ __launch_bounds__(256)
void gemm_proj_kernel(const __bf16* __restrict__ A, const __bf16* __restrict__ BT,
                      float* __restrict__ C) {
  __shared__ __bf16 As[2][128 * 40];
  __shared__ __bf16 Bs[2][128 * 40];

  const int tid  = threadIdx.x;
  const int lane = tid & 31;
  const int w    = tid >> 5;
  const int wr   = w >> 1;
  const int wc   = w & 1;
  const int half = lane >> 4;
  const int l16  = lane & 15;
  const int m0   = blockIdx.y * 128;
  const int n0   = blockIdx.x * 128;
  const int srow = tid >> 1;
  const int spar = tid & 1;

  f32x8 acc[2][4];
#pragma unroll
  for (int i = 0; i < 2; ++i)
#pragma unroll
    for (int j = 0; j < 4; ++j) acc[i][j] = zero8();

  auto stage = [&](int k0, int buf) {
    const unsigned ga = (unsigned)(((m0 + srow) * 1024 + k0 + spar * 16) * 2);
    const unsigned la = lds_addr(&As[buf][srow * 40 + spar * 16]);
    async_b128(la, A, ga);
    async_b128(la + 16, A, ga + 16);
    const unsigned gb = (unsigned)(((n0 + srow) * 1024 + k0 + spar * 16) * 2);
    const unsigned lb = lds_addr(&Bs[buf][srow * 40 + spar * 16]);
    async_b128(lb, BT, gb);
    async_b128(lb + 16, BT, gb + 16);
  };

  stage(0, 0);
  for (int t = 0; t < 32; ++t) {
    const int buf = t & 1;
    if (t < 31) {
      stage((t + 1) * 32, buf ^ 1);
      WAIT_ASYNC(0x4);
    } else {
      WAIT_ASYNC(0x0);
    }
    __syncthreads();

    bf16x16 af[2], bfr[4];
#pragma unroll
    for (int i = 0; i < 2; ++i)
      af[i] = make_afrag(&As[buf][(wr * 32 + i * 16 + l16) * 40], half);
#pragma unroll
    for (int j = 0; j < 4; ++j)
      bfr[j] = make_bfrag(&Bs[buf][(wc * 64 + j * 16 + l16) * 40], half);
#pragma unroll
    for (int i = 0; i < 2; ++i)
#pragma unroll
      for (int j = 0; j < 4; ++j) acc[i][j] = wmma_bf16(af[i], bfr[j], acc[i][j]);
    __syncthreads();
  }

#pragma unroll
  for (int i = 0; i < 2; ++i) {
#pragma unroll
    for (int j = 0; j < 4; ++j) {
      const int n = n0 + wc * 64 + j * 16 + l16;
#pragma unroll
      for (int r = 0; r < 8; ++r) {
        const int m = m0 + wr * 32 + i * 16 + half * 8 + r;
        C[(size_t)m * 1024 + n] = acc[i][j][r];
      }
    }
  }
}

// ---------------------------------------------------------------------------
// Launcher
// ---------------------------------------------------------------------------
extern "C" void kernel_launch(void* const* d_in, const int* in_sizes, int n_in,
                              void* d_out, int out_size, void* d_ws, size_t ws_size,
                              hipStream_t stream) {
  (void)in_sizes; (void)n_in; (void)out_size; (void)ws_size;
  const float* x     = (const float*)d_in[0];
  const float* Wqkv  = (const float*)d_in[1];
  const float* Wproj = (const float*)d_in[2];
  const float* lq1   = (const float*)d_in[3];
  const float* lk1   = (const float*)d_in[4];
  const float* lq2   = (const float*)d_in[5];
  const float* lk2   = (const float*)d_in[6];
  const float* rms   = (const float*)d_in[7];
  float* out = (float*)d_out;

  char* ws = (char*)d_ws;
  size_t off = 0;
  auto alloc = [&](size_t bytes) {
    void* p = ws + off;
    off += (bytes + 255) & ~(size_t)255;
    return p;
  };
  const size_t NE = 2ull * 2048 * 1024;  // 4,194,304 elements
  __bf16* xb     = (__bf16*)alloc(NE * 2);            // x bf16; reused as yb
  __bf16* WqkvT  = (__bf16*)alloc(3ull * 1024 * 1024 * 2);
  __bf16* WprojT = (__bf16*)alloc(1ull * 1024 * 1024 * 2);
  __bf16* qb     = (__bf16*)alloc(NE * 2);
  __bf16* kb     = (__bf16*)alloc(NE * 2);
  __bf16* vtb    = (__bf16*)alloc(NE * 2);
  __bf16* yb     = xb;  // x no longer needed after QKV GEMM

  cvt_f32_bf16<<<(int)((NE + 255) / 256), 256, 0, stream>>>(x, xb, (int)NE);
  cvt_transpose_bf16<<<(3 * 1024 * 1024 + 255) / 256, 256, 0, stream>>>(
      Wqkv, WqkvT, 1024, 3072);
  cvt_transpose_bf16<<<(1024 * 1024 + 255) / 256, 256, 0, stream>>>(
      Wproj, WprojT, 1024, 1024);

  gemm_qkv_kernel<<<dim3(24, 32), 256, 0, stream>>>(xb, WqkvT, qb, kb, vtb);

  diff_attn_kernel<<<dim3(32, 32), 128, 0, stream>>>(qb, kb, vtb, lq1, lk1, lq2,
                                                     lk2, rms, yb);

  gemm_proj_kernel<<<dim3(8, 32), 256, 0, stream>>>(yb, WprojT, out);
}
